// SwinTransformerBlock3D_38920993636681
// MI455X (gfx1250) — matrix operand order repaired and checked by reading, not compiled, not run on hardware
//
#include <hip/hip_runtime.h>
#include <hip/hip_bf16.h>
#include <math.h>

// ---------------------------------------------------------------------------
// Swin-3D block for MI455X (gfx1250): all GEMMs via v_wmma_f32_16x16x32_bf16,
// GEMM tile staging via async direct-to-LDS DMA (ASYNCcnt), double-buffered.
// ---------------------------------------------------------------------------

typedef __attribute__((ext_vector_type(16))) __bf16 v16bf;
typedef __attribute__((ext_vector_type(8)))  float  v8f;

union Frag16 {               // 8 VGPRs = 16 bf16 lanes-worth of A or B operand
    v16bf        v;
    unsigned int u[8];
};

__device__ __forceinline__ unsigned short f2bf(float f) {
    unsigned int u = __float_as_uint(f);
    unsigned int r = u + 0x7FFFu + ((u >> 16) & 1u);   // round-nearest-even
    return (unsigned short)(r >> 16);
}

// Async global->LDS copy, 16 bytes per lane.  GVS addressing:
//   mem  = SADDR(64) + VADDR(32, bytes);  LDS = lds_base + VDST(32, bytes)
// Tracked by ASYNCcnt (CDNA5 ISA).
__device__ __forceinline__ void async_b128(unsigned lds_off, unsigned goff,
                                           const void* base) {
    asm volatile("global_load_async_to_lds_b128 %0, %1, %2"
                 :: "v"(lds_off), "v"(goff), "s"(base)
                 : "memory");
}
__device__ __forceinline__ void wait_async_le3() {
    asm volatile("s_wait_asynccnt 0x3" ::: "memory");
}
__device__ __forceinline__ void wait_async_0() {
    asm volatile("s_wait_asynccnt 0x0" ::: "memory");
}

// ---------------------------------------------------------------------------
// Small helpers
// ---------------------------------------------------------------------------
__global__ __launch_bounds__(256) void to_bf16_k(const float* __restrict__ in,
                                                 unsigned short* __restrict__ out,
                                                 int n) {
    int i = blockIdx.x * 256 + threadIdx.x;
    if (i < n) out[i] = f2bf(in[i]);
}

// biasmat[head][m][n] = bias_table[rel_idx(m,n), head]   (6 x 128 x 128 f32)
__global__ __launch_bounds__(128) void bias_build_k(const float* __restrict__ tab,
                                                    float* __restrict__ bm) {
    int m    = blockIdx.x & 127;
    int head = blockIdx.x >> 7;
    int n    = threadIdx.x;
    int t1 = m >> 6, h1 = (m >> 3) & 7, w1 = m & 7;
    int t2 = n >> 6, h2 = (n >> 3) & 7, w2 = n & 7;
    int idx = (t1 - t2 + 1) * 225 + (h1 - h2 + 7) * 15 + (w1 - w2 + 7);
    bm[((size_t)head * 128 + m) * 128 + n] = tab[idx * 6 + head];
}

// LayerNorm over C=192, one token per wave (lane holds 6 strided channels),
// optional fused window-gather, bf16 output.
template <bool GATHER>
__global__ __launch_bounds__(256) void layernorm_k(const float* __restrict__ xin,
                                                   const float* __restrict__ gamma,
                                                   const float* __restrict__ beta,
                                                   unsigned short* __restrict__ out) {
    const int tid  = threadIdx.x;
    const int wave = tid >> 5;
    const int lane = tid & 31;
    const size_t token = (size_t)blockIdx.x * 8 + wave;
    const float* px = xin + token * 192;

    float v[6];
    float sum = 0.f;
#pragma unroll
    for (int i = 0; i < 6; ++i) { v[i] = px[lane + 32 * i]; sum += v[i]; }
#pragma unroll
    for (int m = 16; m >= 1; m >>= 1) sum += __shfl_xor(sum, m, 32);
    float mean = sum * (1.0f / 192.0f);
    float var = 0.f;
#pragma unroll
    for (int i = 0; i < 6; ++i) { float d = v[i] - mean; var += d * d; }
#pragma unroll
    for (int m = 16; m >= 1; m >>= 1) var += __shfl_xor(var, m, 32);
    float rstd = rsqrtf(var * (1.0f / 192.0f) + 1e-5f);

    size_t orow;
    if (GATHER) {
        int t  = (int)token;
        int w  = t & 127, h = (t >> 7) & 127, tt = (t >> 14) & 7, b = (int)(token >> 17);
        int win = ((b * 4 + (tt >> 1)) * 16 + (h >> 3)) * 16 + (w >> 3);
        int tok = (tt & 1) * 64 + (h & 7) * 8 + (w & 7);
        orow = (size_t)win * 128 + tok;
    } else {
        orow = token;
    }
    unsigned short* po = out + orow * 192;
#pragma unroll
    for (int i = 0; i < 6; ++i) {
        int c = lane + 32 * i;
        po[c] = f2bf((v[i] - mean) * rstd * gamma[c] + beta[c]);
    }
}

// ---------------------------------------------------------------------------
// Generic bf16 WMMA GEMM: C[M,N] = A[M,K] * W[N,K]^T + bias[N], fused epilogue.
// Tile: BM=128 x BN=64, K-step 32, 8 waves (wave w -> rows [16w,16w+16)).
// Tiles staged by async DMA into double-buffered dynamic LDS (2 x 12KB):
//   buf j&1: A tile 128x32 bf16 at +0 (8KB), W tile 64x32 bf16 at +8192 (4KB).
// ---------------------------------------------------------------------------
#define EPI_QKV      0   // split into q/k/v [win][3][head][tok][32] bf16, q*=SCALE
#define EPI_PROJ_RES 1   // un-window-gather, += residual x, f32 out [B,T,H,W,C]
#define EPI_GELU     2   // exact GELU, bf16 out [M,N]
#define EPI_FC2_RES  3   // += residual, f32 out [M,192]

template <int EPI>
__global__ __launch_bounds__(256) void gemm_bf16_k(
        const unsigned short* __restrict__ A,   // [M,K] bf16 row-major
        const unsigned short* __restrict__ W,   // [N,K] bf16 row-major
        const float* __restrict__ bias,         // [N]
        int K, int N,
        unsigned short* __restrict__ outb,      // bf16 destination (QKV / GELU)
        float* __restrict__ outf,               // f32 destination (PROJ / FC2)
        const float* __restrict__ resid)        // residual source (PROJ / FC2)
{
    extern __shared__ char smem[];              // 24KB dynamic (LDS offset 0)

    const int tid  = threadIdx.x;
    const int wave = tid >> 5;
    const int lane = tid & 31;
    const int half = lane >> 4;
    const int l16  = lane & 15;
    const int m0   = blockIdx.x * 128;
    const int n0   = blockIdx.y * 64;

    // Per-thread staging coordinates (A: 2 x b128, W: 1 x b128 per K-step)
    const int arow = tid >> 1, acb = (tid & 1) * 16;
    const int wrow = tid >> 2, wcb = (tid & 3) * 8;

    const int nk = K >> 5;
    v8f acc[4] = {};

    auto issue = [&](int j) {
        unsigned kb  = (unsigned)j << 5;
        unsigned buf = (unsigned)(j & 1) * 12288u;
        unsigned ga  = ((unsigned)(m0 + arow) * (unsigned)K + kb + (unsigned)acb) * 2u;
        unsigned la  = buf + (unsigned)(arow * 32 + acb) * 2u;
        async_b128(la,       ga,       A);
        async_b128(la + 16u, ga + 16u, A);
        unsigned gw  = ((unsigned)(n0 + wrow) * (unsigned)K + kb + (unsigned)wcb) * 2u;
        unsigned lw  = buf + 8192u + (unsigned)(wrow * 32 + wcb) * 2u;
        async_b128(lw, gw, W);
    };

    issue(0);
    for (int i = 0; i < nk; ++i) {
        if (i + 1 < nk) { issue(i + 1); wait_async_le3(); }   // batch i retired
        else            {               wait_async_0();   }
        __syncthreads();

        const unsigned short* lA =
            (const unsigned short*)(smem + (size_t)(i & 1) * 12288);
        const unsigned short* lW = lA + 4096;

        // A fragment (16x32 bf16): lane L -> M=L%16; VGPR v -> K = half*16 + 2v
        Frag16 fa;
        {
            const unsigned int* p =
                (const unsigned int*)(lA + (wave * 16 + l16) * 32);
#pragma unroll
            for (int v = 0; v < 8; ++v) {
                int kk = (half << 4) + (v << 1);
                fa.u[v] = p[kk >> 1];
            }
        }
#pragma unroll
        for (int nf = 0; nf < 4; ++nf) {
            // B fragment (32x16 bf16): lane L -> N=L%16; K = half*16 + 2v
            Frag16 fb;
            const unsigned int* p =
                (const unsigned int*)(lW + (nf * 16 + l16) * 32 + (half << 4));
#pragma unroll
            for (int v = 0; v < 8; ++v) fb.u[v] = p[v];
            acc[nf] = __builtin_amdgcn_wmma_f32_16x16x32_bf16(
                false, fa.v, false, fb.v, (short)0, acc[nf], false, false);
        }
        __syncthreads();   // everyone done reading buf(i&1) before its refill
    }

    // Epilogue.  C/D layout: lane L, VGPR v -> M_local = v + 8*(L>>4), N_local = L%16
#pragma unroll
    for (int nf = 0; nf < 4; ++nf) {
#pragma unroll
        for (int v = 0; v < 8; ++v) {
            int m = m0 + wave * 16 + v + half * 8;
            int n = n0 + nf * 16 + l16;
            float val = acc[nf][v] + bias[n];

            if (EPI == EPI_QKV) {
                int g = n / 192;             // 0=q 1=k 2=v
                int r = n - g * 192;
                int head = r >> 5;
                int d = r & 31;
                if (g == 0) val *= 0.17677669529663687f;  // head_dim^-0.5
                int win = m >> 7, tok = m & 127;
                outb[((((size_t)win * 3 + g) * 6 + head) * 128 + tok) * 32 + d] =
                    f2bf(val);
            } else if (EPI == EPI_PROJ_RES) {
                int win = m >> 7, tok = m & 127;
                int wb = win & 15, hb = (win >> 4) & 15, tb = (win >> 8) & 3;
                int b = win >> 10;
                int t = tb * 2 + (tok >> 6);
                int h = hb * 8 + ((tok >> 3) & 7);
                int w = wb * 8 + (tok & 7);
                size_t orig = (((size_t)b * 8 + t) * 128 + h) * 128 + w;
                outf[orig * 192 + n] = val + resid[orig * 192 + n];
            } else if (EPI == EPI_GELU) {
                float gl = 0.5f * val * (1.0f + erff(val * 0.70710678118654752f));
                outb[(size_t)m * N + n] = f2bf(gl);
            } else {  // EPI_FC2_RES
                outf[(size_t)m * 192 + n] = resid[(size_t)m * 192 + n] + val;
            }
        }
    }
}

// ---------------------------------------------------------------------------
// Windowed attention: one block per (window, head); 8 waves, each owns a
// 16x128 score slab. q*kT (8 WMMA) -> +bias -> softmax in registers ->
// P through LDS -> P*V (8 WMMA) -> bf16 out [win*128+tok, head*32+d].
// ---------------------------------------------------------------------------
__global__ __launch_bounds__(256) void attention_k(
        const unsigned short* __restrict__ qkv,   // [win][3][6][128][32] bf16
        const float* __restrict__ biasmat,        // [6][128][128] f32
        unsigned short* __restrict__ out)         // [win*128, 192] bf16
{
    __shared__ __attribute__((aligned(16))) unsigned short lP[8][16 * 128]; // 32KB
    __shared__ __attribute__((aligned(16))) unsigned short lVt[32 * 128];   // 8KB

    const int blk  = blockIdx.x;
    const int win  = blk / 6;
    const int head = blk - win * 6;
    const unsigned short* qb = qkv + ((size_t)(win * 3 + 0) * 6 + head) * 4096;
    const unsigned short* kb = qkv + ((size_t)(win * 3 + 1) * 6 + head) * 4096;
    const unsigned short* vb = qkv + ((size_t)(win * 3 + 2) * 6 + head) * 4096;

    const int tid  = threadIdx.x;
    const int wave = tid >> 5;
    const int lane = tid & 31;
    const int half = lane >> 4;
    const int l16  = lane & 15;

    // Stage V transposed: lVt[d][tok]  (so P*V B-fragments are contiguous in K)
    for (int i = tid; i < 4096; i += 256) {
        int tok = i >> 5, d = i & 31;
        lVt[d * 128 + tok] = vb[i];
    }
    __syncthreads();

    // ---- S = (q*scale) kT : A = q rows [16w..16w+16) x 32, K = head_dim ----
    Frag16 fa;
    {
        const unsigned int* qp = (const unsigned int*)(qb + (wave * 16 + l16) * 32);
#pragma unroll
        for (int v = 0; v < 8; ++v) {
            int kk = (half << 4) + (v << 1);
            fa.u[v] = qp[kk >> 1];
        }
    }
    v8f s[8];
#pragma unroll
    for (int nt = 0; nt < 8; ++nt) {
        Frag16 fb;  // B col n = key token nt*16+l16, K = d
        const unsigned int* kp =
            (const unsigned int*)(kb + (nt * 16 + l16) * 32 + (half << 4));
#pragma unroll
        for (int v = 0; v < 8; ++v) fb.u[v] = kp[v];
        v8f z = {};
        s[nt] = __builtin_amdgcn_wmma_f32_16x16x32_bf16(
            false, fa.v, false, fb.v, (short)0, z, false, false);
    }

    // ---- relative-position bias ----
    const float* bm = biasmat + (size_t)head * 128 * 128;
#pragma unroll
    for (int nt = 0; nt < 8; ++nt) {
#pragma unroll
        for (int v = 0; v < 8; ++v) {
            int m = wave * 16 + v + half * 8;
            int n = nt * 16 + l16;
            s[nt][v] += bm[m * 128 + n];
        }
    }

    // ---- softmax over the 128 columns of each row (16-lane half reductions) ----
#pragma unroll
    for (int v = 0; v < 8; ++v) {
        float mx = -1e30f;
#pragma unroll
        for (int nt = 0; nt < 8; ++nt) mx = fmaxf(mx, s[nt][v]);
#pragma unroll
        for (int mm = 8; mm >= 1; mm >>= 1) mx = fmaxf(mx, __shfl_xor(mx, mm, 32));
        float sm = 0.f;
#pragma unroll
        for (int nt = 0; nt < 8; ++nt) {
            float e = __expf(s[nt][v] - mx);
            s[nt][v] = e;
            sm += e;
        }
#pragma unroll
        for (int mm = 8; mm >= 1; mm >>= 1) sm += __shfl_xor(sm, mm, 32);
        float inv = 1.0f / sm;
#pragma unroll
        for (int nt = 0; nt < 8; ++nt) s[nt][v] *= inv;
    }

    // ---- P -> LDS (per-wave private slab; same-wave RAW handled by dscnt) ----
    unsigned short* P = &lP[wave][0];
#pragma unroll
    for (int nt = 0; nt < 8; ++nt) {
#pragma unroll
        for (int v = 0; v < 8; ++v) {
            int ml = v + half * 8;
            P[ml * 128 + nt * 16 + l16] = f2bf(s[nt][v]);
        }
    }

    // ---- O = P (16x128) * V (128x32): 4 K-steps x 2 N-tiles ----
    v8f o[2] = {};
#pragma unroll
    for (int kk = 0; kk < 4; ++kk) {
        Frag16 pa;
        const unsigned int* pp =
            (const unsigned int*)(P + l16 * 128 + kk * 32);
#pragma unroll
        for (int v = 0; v < 8; ++v) {
            int k2 = (half << 4) + (v << 1);
            pa.u[v] = pp[k2 >> 1];
        }
#pragma unroll
        for (int nt2 = 0; nt2 < 2; ++nt2) {
            Frag16 fv;  // B col = d, K rows = token kk*32 + half*16 + 2v
            const unsigned int* vp = (const unsigned int*)(
                lVt + (nt2 * 16 + l16) * 128 + kk * 32 + (half << 4));
#pragma unroll
            for (int v = 0; v < 8; ++v) fv.u[v] = vp[v];
            o[nt2] = __builtin_amdgcn_wmma_f32_16x16x32_bf16(
                false, pa.v, false, fv.v, (short)0, o[nt2], false, false);
        }
    }

    // ---- store [win*128+tok, head*32+d] bf16 ----
    size_t rowbase = ((size_t)win * 128 + wave * 16) * 192;
#pragma unroll
    for (int nt2 = 0; nt2 < 2; ++nt2) {
#pragma unroll
        for (int v = 0; v < 8; ++v) {
            int ml = v + half * 8;
            int ch = head * 32 + nt2 * 16 + l16;
            out[rowbase + (size_t)ml * 192 + ch] = f2bf(o[nt2][v]);
        }
    }
}

// ---------------------------------------------------------------------------
// Launch
// ---------------------------------------------------------------------------
extern "C" void kernel_launch(void* const* d_in, const int* in_sizes, int n_in,
                              void* d_out, int out_size, void* d_ws, size_t ws_size,
                              hipStream_t stream) {
    (void)in_sizes; (void)n_in; (void)out_size; (void)ws_size;

    const float* x     = (const float*)d_in[0];
    const float* n1g   = (const float*)d_in[1];
    const float* n1b   = (const float*)d_in[2];
    const float* qkvw  = (const float*)d_in[3];
    const float* qkvb  = (const float*)d_in[4];
    const float* btab  = (const float*)d_in[5];
    const float* projw = (const float*)d_in[6];
    const float* projb = (const float*)d_in[7];
    const float* n2g   = (const float*)d_in[8];
    const float* n2b   = (const float*)d_in[9];
    const float* f1w   = (const float*)d_in[10];
    const float* f1b   = (const float*)d_in[11];
    const float* f2w   = (const float*)d_in[12];
    const float* f2b   = (const float*)d_in[13];

    char* ws = (char*)d_ws;
    // bf16 weights + bias matrix
    unsigned short* wq  = (unsigned short*)(ws + 0);          // 576*192
    unsigned short* wp  = (unsigned short*)(ws + 221184);     // 192*192
    unsigned short* wf1 = (unsigned short*)(ws + 294912);     // 768*192
    unsigned short* wf2 = (unsigned short*)(ws + 589824);     // 192*768
    float*          bm  = (float*)(ws + 884736);              // 6*128*128 f32
    // Big buffers (regions reused across pipeline stages)
    const size_t o_xw  = 2097152;                  // 96MB : Xw bf16 / attn_out bf16
    const size_t o_qkv = o_xw + 100663296;         // 288MB: qkv bf16 / X1 f32
    const size_t o_x1n = o_qkv + 301989888;        // 96MB : X1n bf16
    const size_t o_h1  = o_x1n + 100663296;        // 384MB: H1 bf16
    unsigned short* xw   = (unsigned short*)(ws + o_xw);
    unsigned short* qkv  = (unsigned short*)(ws + o_qkv);
    unsigned short* attn = (unsigned short*)(ws + o_xw);   // reuse Xw
    float*          x1   = (float*)(ws + o_qkv);           // reuse qkv
    unsigned short* x1n  = (unsigned short*)(ws + o_x1n);
    unsigned short* h1   = (unsigned short*)(ws + o_h1);
    float*          y    = (float*)d_out;

    // Weight conversion + bias table
    to_bf16_k<<<110592 / 256, 256, 0, stream>>>(qkvw, wq, 110592);
    to_bf16_k<<<36864 / 256, 256, 0, stream>>>(projw, wp, 36864);
    to_bf16_k<<<147456 / 256, 256, 0, stream>>>(f1w, wf1, 147456);
    to_bf16_k<<<147456 / 256, 256, 0, stream>>>(f2w, wf2, 147456);
    bias_build_k<<<6 * 128, 128, 0, stream>>>(btab, bm);

    // LN1 + window gather (262144 tokens, 8/block)
    layernorm_k<true><<<32768, 256, 0, stream>>>(x, n1g, n1b, xw);

    const size_t gemm_lds = 24576;   // 2 x (8KB A + 4KB W) double buffer

    // QKV GEMM: M=262144, N=576, K=192
    gemm_bf16_k<EPI_QKV><<<dim3(2048, 9), 256, gemm_lds, stream>>>(
        xw, wq, qkvb, 192, 576, qkv, nullptr, nullptr);

    // Attention: 2048 windows x 6 heads
    attention_k<<<2048 * 6, 256, 0, stream>>>(qkv, bm, attn);

    // Proj GEMM + un-gather + residual: M=262144, N=192, K=192 -> X1 f32
    gemm_bf16_k<EPI_PROJ_RES><<<dim3(2048, 3), 256, gemm_lds, stream>>>(
        attn, wp, projb, 192, 192, nullptr, x1, x);

    // LN2 (flat)
    layernorm_k<false><<<32768, 256, 0, stream>>>(x1, n2g, n2b, x1n);

    // FC1 + GELU: M=262144, N=768, K=192 -> H1 bf16
    gemm_bf16_k<EPI_GELU><<<dim3(2048, 12), 256, gemm_lds, stream>>>(
        x1n, wf1, f1b, 192, 768, h1, nullptr, nullptr);

    // FC2 + residual: M=262144, N=192, K=768 -> d_out f32
    gemm_bf16_k<EPI_FC2_RES><<<dim3(2048, 3), 256, gemm_lds, stream>>>(
        h1, wf2, f2b, 768, 192, nullptr, y, x1);
}
